// FeatureLoss_50861002719940
// MI455X (gfx1250) — 8-bit matrix operand order repaired and checked by reading, compiled, not run on hardware
//
#include <hip/hip_runtime.h>
#include <hip/hip_bf16.h>

#define FEAT  2048
#define KROWS 4096
#define BDIM  8192
#define EPSV  1e-12f

typedef int   v8i __attribute__((ext_vector_type(8)));
typedef float v8f __attribute__((ext_vector_type(8)));

#define WS_NORM2  0
#define WS_SUMEXP (4096*4)
#define WS_POS    (2*4096*4)
#define WS_ENT    (3*4096*4)
#define WS_CE     (3*4096*4 + 4)
#define WS_FNQ    65536
#define WS_NEED   (WS_FNQ + (size_t)KROWS*BDIM)

#if defined(__has_builtin)
#if __has_builtin(__builtin_amdgcn_cvt_pk_fp8_f32)
#define HAVE_PK_FP8 1
#endif
#endif

__device__ __forceinline__ unsigned int sw_e4m3(float x) {
  x = fminf(fmaxf(x, 0.0f), 448.0f);
  if (x < 9.765625e-4f) return 0u;
  unsigned u = __float_as_uint(x);
  int e = (int)((u >> 23) & 255) - 127;
  unsigned m = u & 0x7fffffu;
  if (e < -6) {
    int shift = -6 - e;
    m = (m | 0x800000u) >> shift;
    unsigned r = (m + (1u << 19)) >> 20;
    if (r > 7u) r = 7u;
    return r;
  }
  unsigned mant = (m + (1u << 19)) >> 20;
  if (mant == 8u) { mant = 0u; e += 1; }
  if (e > 8) { e = 8; mant = 6u; }
  return (((unsigned)(e + 7)) << 3) | mant;
}

__device__ __forceinline__ unsigned int pack4_e4m3(float a, float b, float c, float d) {
#ifdef HAVE_PK_FP8
  unsigned int r = (unsigned int)__builtin_amdgcn_cvt_pk_fp8_f32(a, b, 0, false);
  r = (unsigned int)__builtin_amdgcn_cvt_pk_fp8_f32(c, d, (int)r, true);
  return r;
#else
  return sw_e4m3(a) | (sw_e4m3(b) << 8) | (sw_e4m3(c) << 16) | (sw_e4m3(d) << 24);
#endif
}

__global__ void __launch_bounds__(256)
k_norm_ent(const float* __restrict__ f1, const float* __restrict__ f2,
           float* __restrict__ norm2, float* __restrict__ ent_sum) {
  const int c  = blockIdx.x * 256 + threadIdx.x;
  const int b0 = blockIdx.y * 1024;
  const int z  = blockIdx.z;
  const float* __restrict__ src = z ? f2 : f1;
  float ss = 0.0f, le = 0.0f;
#pragma unroll 4
  for (int i = 0; i < 1024; ++i) {
    float x = src[(size_t)(b0 + i) * FEAT + c];
    ss += x * x;
    le += -x * __logf(x + EPSV) - (1.0f - x) * __logf(1.0f - x + EPSV);
  }
  atomicAdd(&norm2[z * FEAT + c], ss);
  __shared__ float red[256];
  red[threadIdx.x] = le;
  __syncthreads();
  for (int s = 128; s > 0; s >>= 1) {
    if (threadIdx.x < s) red[threadIdx.x] += red[threadIdx.x + s];
    __syncthreads();
  }
  if (threadIdx.x == 0) atomicAdd(ent_sum, red[0]);
}

__global__ void __launch_bounds__(256)
k_quant(const float* __restrict__ f1, const float* __restrict__ f2,
        const float* __restrict__ norm2, unsigned char* __restrict__ fnq) {
  __shared__ float tile[32][132];
  const int tx = threadIdx.x;
  const int ty = threadIdx.y;
  const int j0 = blockIdx.x * 32;
  const int b0 = blockIdx.y * 128;
  const int z  = blockIdx.z;
  const float* __restrict__ src = z ? f2 : f1;
#pragma unroll
  for (int i = 0; i < 16; ++i) {
    int br = ty + 8 * i;
    tile[tx][br] = src[(size_t)(b0 + br) * FEAT + j0 + tx];
  }
  __syncthreads();
#pragma unroll
  for (int i = 0; i < 4; ++i) {
    int kk  = ty + 8 * i;
    int row = z * FEAT + j0 + kk;
    float s = 64.0f / fmaxf(sqrtf(norm2[row]), EPSV);
    float a = tile[kk][4 * tx + 0] * s;
    float b = tile[kk][4 * tx + 1] * s;
    float c = tile[kk][4 * tx + 2] * s;
    float d = tile[kk][4 * tx + 3] * s;
    *(unsigned int*)(fnq + (size_t)row * BDIM + b0 + 4 * tx) = pack4_e4m3(a, b, c, d);
  }
}

#define KP      128
#define COLPAD  144
#define NPANEL  (BDIM / KP)

__global__ void __launch_bounds__(256)
k_gemm(const unsigned char* __restrict__ fnq,
       float* __restrict__ sumexp, float* __restrict__ posv) {
  __shared__ __align__(16) unsigned char ldsB[2][128 * COLPAD];

  const int tid  = threadIdx.x;
  const int wv   = tid >> 5;
  const int lane = tid & 31;
  const bool lo  = lane < 16;
  const int ln   = lane & 15;
  const int m0   = blockIdx.x * 128 + wv * 16;
  const int n0   = blockIdx.y * 128;

  const unsigned char* rowA = fnq + (size_t)(m0 + ln) * BDIM;

  v8f acc[8];
#pragma unroll
  for (int t = 0; t < 8; ++t)
#pragma unroll
    for (int r = 0; r < 8; ++r) acc[t][r] = 0.0f;

  const int aoff0 = lo ? 0 : 16;
  const int aoff1 = lo ? 8 : 24;
  const int boff0 = lo ? 0 : 16;

  const int scol0 = tid >> 3;
  const int sko   = (tid & 7) * 16;

  {
    const unsigned char* g = fnq + (size_t)(n0 + scol0) * BDIM + sko;
#pragma unroll
    for (int i = 0; i < 4; ++i) {
      unsigned lds = (unsigned)(size_t)&ldsB[0][(scol0 + 32 * i) * COLPAD + sko];
      const unsigned char* gp = g + (size_t)(32 * i) * BDIM;
      asm volatile("global_load_async_to_lds_b128 %0, %1, off"
                   :: "v"(lds), "v"(gp) : "memory");
    }
  }

  for (int p = 0; p < NPANEL; ++p) {
    asm volatile("s_wait_asynccnt 0x0" ::: "memory");
    __syncthreads();

    if (p + 1 < NPANEL) {
      const int nb = (p + 1) & 1;
      const unsigned char* g = fnq + (size_t)(n0 + scol0) * BDIM + (p + 1) * KP + sko;
#pragma unroll
      for (int i = 0; i < 4; ++i) {
        unsigned lds = (unsigned)(size_t)&ldsB[nb][(scol0 + 32 * i) * COLPAD + sko];
        const unsigned char* gp = g + (size_t)(32 * i) * BDIM;
        asm volatile("global_load_async_to_lds_b128 %0, %1, off"
                     :: "v"(lds), "v"(gp) : "memory");
      }
    }

    const unsigned char* bb = &ldsB[p & 1][0];
#pragma unroll
    for (int s = 0; s < 2; ++s) {
      const int b0 = p * KP + s * 64;
      union { v8i v; unsigned long long d[4]; } A;
      A.d[0] = *(const unsigned long long*)(rowA + b0 + aoff0);
      A.d[1] = *(const unsigned long long*)(rowA + b0 + aoff1);
      A.d[2] = *(const unsigned long long*)(rowA + b0 + aoff0 + 32);
      A.d[3] = *(const unsigned long long*)(rowA + b0 + aoff1 + 32);
#pragma unroll
      for (int t = 0; t < 8; ++t) {
        const unsigned char* bc = bb + (t * 16 + ln) * COLPAD + s * 64 + boff0;
        union { v8i v; int4 q[2]; } Bf;
        Bf.q[0] = *(const int4*)(bc);
        Bf.q[1] = *(const int4*)(bc + 32);
        acc[t] = __builtin_amdgcn_wmma_f32_16x16x64_fp8_fp8(A.v, Bf.v, (short)0, acc[t],
                                                            false, false);
      }
    }
  }

  const float post = 1.0f / (64.0f * 64.0f * 0.5f);
  float rs[8];
#pragma unroll
  for (int r = 0; r < 8; ++r) rs[r] = 0.0f;
#pragma unroll
  for (int t = 0; t < 8; ++t) {
    int col = n0 + t * 16 + ln;
#pragma unroll
    for (int r = 0; r < 8; ++r) {
      int row   = m0 + r + (lo ? 0 : 8);
      float val = acc[t][r] * post;
      if (col == (row ^ FEAT)) posv[row] = val;
      if (col != row) rs[r] += __expf(val);
    }
  }
#pragma unroll
  for (int r = 0; r < 8; ++r) {
    float v = rs[r];
    v += __shfl_xor(v, 8);
    v += __shfl_xor(v, 4);
    v += __shfl_xor(v, 2);
    v += __shfl_xor(v, 1);
    if (ln == 0) {
      int row = m0 + r + (lo ? 0 : 8);
      atomicAdd(&sumexp[row], v);
    }
  }
}

__global__ void __launch_bounds__(256)
k_rows(const float* __restrict__ sumexp, const float* __restrict__ posv,
       float* __restrict__ ce_sum) {
  int k = blockIdx.x * 256 + threadIdx.x;
  float contrib = __logf(sumexp[k]) - posv[k];
  __shared__ float red[256];
  red[threadIdx.x] = contrib;
  __syncthreads();
  for (int s = 128; s > 0; s >>= 1) {
    if (threadIdx.x < s) red[threadIdx.x] += red[threadIdx.x + s];
    __syncthreads();
  }
  if (threadIdx.x == 0) atomicAdd(ce_sum, red[0]);
}

__global__ void k_final(const float* __restrict__ ce_sum,
                        const float* __restrict__ ent_sum, float* __restrict__ out) {
  out[0] = ce_sum[0] * (1.0f / (float)KROWS)
         - ent_sum[0] * (1.0f / ((float)KROWS * (float)BDIM * 0.69314718055994531f));
}

extern "C" void kernel_launch(void* const* d_in, const int* in_sizes, int n_in,
                              void* d_out, int out_size, void* d_ws, size_t ws_size,
                              hipStream_t stream) {
  (void)in_sizes; (void)n_in; (void)out_size;
  if (ws_size < WS_NEED) return;

  const float* f1 = (const float*)d_in[0];
  const float* f2 = (const float*)d_in[1];
  float* out = (float*)d_out;
  char*  ws  = (char*)d_ws;
  float* norm2  = (float*)(ws + WS_NORM2);
  float* sumexp = (float*)(ws + WS_SUMEXP);
  float* posv   = (float*)(ws + WS_POS);
  float* ent    = (float*)(ws + WS_ENT);
  float* ce     = (float*)(ws + WS_CE);
  unsigned char* fnq = (unsigned char*)(ws + WS_FNQ);

  hipMemsetAsync(ws, 0, 65536, stream);

  k_norm_ent<<<dim3(FEAT / 256, BDIM / 1024, 2), 256, 0, stream>>>(f1, f2, norm2, ent);
  k_quant<<<dim3(FEAT / 32, BDIM / 128, 2), dim3(32, 8), 0, stream>>>(f1, f2, norm2, fnq);
  k_gemm<<<dim3(KROWS / 128, KROWS / 128), 256, 0, stream>>>(fnq, sumexp, posv);
  k_rows<<<KROWS / 256, 256, 0, stream>>>(sumexp, posv, ce);
  k_final<<<1, 1, 0, stream>>>(ce, ent, out);
}
